// TileMSEMatcher_6038724018306
// MI455X (gfx1250) — matrix operand order repaired and checked by reading, not compiled, not run on hardware
//
#include <hip/hip_runtime.h>

typedef _Float16 v8h  __attribute__((ext_vector_type(8)));
typedef _Float16 v16h __attribute__((ext_vector_type(16)));
typedef float v2f __attribute__((ext_vector_type(2)));
typedef float v4f __attribute__((ext_vector_type(4)));
typedef float v8f __attribute__((ext_vector_type(8)));

#define KDIM      8
#define BLK       64          // K*K
#define NB        2048        // NBLOCKS
#define IMG_H     1024
#define IMG_W     1024
#define PATW      128         // patches per image row (W/K)
#define LDIM      (128 * 128) // 16384 patches per batch
#define IMG_ELEMS (2 * IMG_H * IMG_W)   // 2097152
#define BLK_ELEMS (NB * BLK)            // 131072
#define NT_PER_WAVE 8         // 16-col N-tiles per wave (128 cols)
#define NGROUPS   (NB / (16 * NT_PER_WAVE)) // 16
#define LTILES    (LDIM / 16)               // 1024
#define WAVES_PER_BLOCK 8

union frag16 { v16h v; v8h h8[2]; };

// ---------------------------------------------------------------------------
// Pass 1a: sqb[n] = sum_k blocks[n,k]^2
// ---------------------------------------------------------------------------
__global__ __launch_bounds__(256)
void sqb_kernel(const float* __restrict__ blocks, float* __restrict__ sqb) {
    const int n = blockIdx.x * blockDim.x + threadIdx.x;   // 0..2047
    const v4f* p = (const v4f*)(blocks + (size_t)n * BLK);
    float s = 0.f;
#pragma unroll
    for (int i = 0; i < BLK / 4; ++i) {
        v4f q = p[i];
        s += q.x * q.x + q.y * q.y + q.z * q.z + q.w * q.w;
    }
    sqb[n] = s;
}

// ---------------------------------------------------------------------------
// Pass 1b: sqa[b,l] = sum over 8x8 patch of img^2
// ---------------------------------------------------------------------------
__global__ __launch_bounds__(256)
void sqa_kernel(const float* __restrict__ img, float* __restrict__ sqa) {
    const int tid = blockIdx.x * blockDim.x + threadIdx.x;  // 0..32767
    const int b  = tid >> 14;
    const int l  = tid & (LDIM - 1);
    const int ph = l >> 7;
    const int pw = l & (PATW - 1);
    const float* base = img + ((size_t)b * IMG_H + (size_t)ph * KDIM) * IMG_W
                            + (size_t)pw * KDIM;
    float s = 0.f;
#pragma unroll
    for (int r = 0; r < KDIM; ++r) {
        v4f q0 = *(const v4f*)(base + (size_t)r * IMG_W);
        v4f q1 = *(const v4f*)(base + (size_t)r * IMG_W + 4);
        s += q0.x * q0.x + q0.y * q0.y + q0.z * q0.z + q0.w * q0.w;
        s += q1.x * q1.x + q1.y * q1.y + q1.z * q1.z + q1.w * q1.w;
    }
    sqa[tid] = s;
}

// ---------------------------------------------------------------------------
// Pass 1c: split blocks (scaled by -1/32) into f16 hi/lo pair.
//   v = -blocks/32 ; bh = f16(v) ; bl = f16(v - bh)   (exact residual)
// ---------------------------------------------------------------------------
__global__ __launch_bounds__(256)
void split_blocks_kernel(const float* __restrict__ blocks,
                         _Float16* __restrict__ bh, _Float16* __restrict__ bl) {
    const int i = blockIdx.x * blockDim.x + threadIdx.x;   // 0..131071
    const float v = blocks[i] * (-1.f / 32.f);
    const _Float16 hi = (_Float16)v;
    bh[i] = hi;
    bl[i] = (_Float16)(v - (float)hi);
}

// ---------------------------------------------------------------------------
// Pass 1d: split image into f16 hi/lo pair (unscaled).
// ---------------------------------------------------------------------------
__global__ __launch_bounds__(256)
void split_img_kernel(const float* __restrict__ img,
                      _Float16* __restrict__ ah, _Float16* __restrict__ al) {
    const int i = blockIdx.x * blockDim.x + threadIdx.x;   // 0..2097151
    const float v = img[i];
    const _Float16 hi = (_Float16)v;
    ah[i] = hi;
    al[i] = (_Float16)(v - (float)hi);
}

// ---------------------------------------------------------------------------
// Pass 2: mse[b,l,n] via split-f16 WMMA (Ah*Bh + Ah*Bl + Al*Bh, f32 accum).
//   B pre-scaled by -1/32; C initialized to (sqa+sqb)/64, so D = mse.
// ---------------------------------------------------------------------------
__global__ __launch_bounds__(256)
void tile_mse_wmma_kernel(const _Float16* __restrict__ ah_ws, // [2,1024,1024]
                          const _Float16* __restrict__ al_ws,
                          const _Float16* __restrict__ bh_ws, // [2048,64]
                          const _Float16* __restrict__ bl_ws,
                          const float* __restrict__ sqb_ws,   // [2048]
                          const float* __restrict__ sqa_ws,   // [2*16384]
                          float* __restrict__ out)            // [2, L, NB]
{
    const int lane = threadIdx.x & 31;
    const int wave = threadIdx.x >> 5;
    const int h    = lane >> 4;   // half-wave: 0 or 1
    const int m    = lane & 15;   // A row within tile / B column within tile

    const int wid = blockIdx.x * WAVES_PER_BLOCK + wave;
    const int b   = wid / (LTILES * NGROUPS);
    const int rem = wid % (LTILES * NGROUPS);
    const int lt  = rem / NGROUPS;   // 16-row L tile
    const int ng  = rem % NGROUPS;   // 128-col N group

    const int ph  = lt >> 3;               // (lt*16)/128
    const int pw0 = (lt & 7) << 4;         // (lt*16)%128
    const int pw  = pw0 + m;

    // A fragments: contiguous k order, matching B. Lane (h,m) holds, for frag f:
    //   halves 0..7  = K(32f + 16h .. +7)     -> patch row kh = 4f + 2h
    //   halves 8..15 = K(32f + 16h + 8 .. +7) -> patch row kh = 4f + 2h + 1
    const size_t arow0 = ((size_t)b * IMG_H + (size_t)ph * KDIM) * IMG_W
                       + (size_t)pw * KDIM;
    frag16 ah[2], al[2];
#pragma unroll
    for (int f = 0; f < 2; ++f) {
        const size_t r0 = arow0 + (size_t)(4 * f + 2 * h) * IMG_W;
        const size_t r1 = arow0 + (size_t)(4 * f + 2 * h + 1) * IMG_W;
        ah[f].h8[0] = *(const v8h*)(ah_ws + r0);
        ah[f].h8[1] = *(const v8h*)(ah_ws + r1);
        al[f].h8[0] = *(const v8h*)(al_ws + r0);
        al[f].h8[1] = *(const v8h*)(al_ws + r1);
    }

    // sqa/64 for output rows (r + 8h): 8 contiguous floats, two b128 loads.
    const float* sqa_base = sqa_ws + (size_t)b * LDIM + (size_t)lt * 16 + 8 * h;
    v4f qa0 = *(const v4f*)(sqa_base);
    v4f qa1 = *(const v4f*)(sqa_base + 4);
    const float cscale = 1.f / 64.f;
    float sqa_s[8];
    sqa_s[0] = qa0.x * cscale; sqa_s[1] = qa0.y * cscale;
    sqa_s[2] = qa0.z * cscale; sqa_s[3] = qa0.w * cscale;
    sqa_s[4] = qa1.x * cscale; sqa_s[5] = qa1.y * cscale;
    sqa_s[6] = qa1.z * cscale; sqa_s[7] = qa1.w * cscale;

    const int n0base = ng * (NT_PER_WAVE * 16);
    float* outb = out + ((size_t)b * LDIM + (size_t)lt * 16) * NB;

#pragma unroll
    for (int nt = 0; nt < NT_PER_WAVE; ++nt) {
        const int n0 = n0base + nt * 16;
        // B fragments: contiguous k order, matching A. Lane (h,m) holds column n0+m,
        // K(32f + 16h .. 16h+15): 16 contiguous halves of blocks row n0+m.
        const size_t boff = (size_t)(n0 + m) * BLK + 16 * h;
        if (nt + 1 < NT_PER_WAVE)
            __builtin_prefetch(bh_ws + boff + 16 * BLK, 0, 3);

        frag16 bh[2], bl[2];
#pragma unroll
        for (int f = 0; f < 2; ++f) {
            bh[f].h8[0] = *(const v8h*)(bh_ws + boff + 32 * f);
            bh[f].h8[1] = *(const v8h*)(bh_ws + boff + 32 * f + 8);
            bl[f].h8[0] = *(const v8h*)(bl_ws + boff + 32 * f);
            bl[f].h8[1] = *(const v8h*)(bl_ws + boff + 32 * f + 8);
        }

        // C init: (sqa + sqb) / 64
        const float sqb_s = sqb_ws[n0 + m] * cscale;
        v8f c;
#pragma unroll
        for (int r = 0; r < 8; ++r) c[r] = sqa_s[r] + sqb_s;

        // D = Ah*Bh + Ah*Bl + Al*Bh + C   (each term: K=64 -> 2 WMMAs)
        c = __builtin_amdgcn_wmma_f32_16x16x32_f16(false, ah[0].v, false, bh[0].v, (short)0, c, false, false);
        c = __builtin_amdgcn_wmma_f32_16x16x32_f16(false, ah[1].v, false, bh[1].v, (short)0, c, false, false);
        c = __builtin_amdgcn_wmma_f32_16x16x32_f16(false, ah[0].v, false, bl[0].v, (short)0, c, false, false);
        c = __builtin_amdgcn_wmma_f32_16x16x32_f16(false, ah[1].v, false, bl[1].v, (short)0, c, false, false);
        c = __builtin_amdgcn_wmma_f32_16x16x32_f16(false, al[0].v, false, bh[0].v, (short)0, c, false, false);
        c = __builtin_amdgcn_wmma_f32_16x16x32_f16(false, al[1].v, false, bh[1].v, (short)0, c, false, false);

        // D *is* the mse tile: store directly.
#pragma unroll
        for (int r = 0; r < 8; ++r)
            outb[(size_t)(r + 8 * h) * NB + (size_t)(n0 + m)] = c[r];
    }
}

extern "C" void kernel_launch(void* const* d_in, const int* in_sizes, int n_in,
                              void* d_out, int out_size, void* d_ws, size_t ws_size,
                              hipStream_t stream) {
    (void)in_sizes; (void)n_in; (void)out_size; (void)ws_size;
    const float* blocks = (const float*)d_in[0];   // [2048, 64] fp32
    const float* img    = (const float*)d_in[1];   // [2, 1, 1024, 1024] fp32
    float* out          = (float*)d_out;           // [2, 16384, 2048] fp32

    // Workspace layout (needs ~9.0 MB):
    //   sqb  f32[2048]      @ 0
    //   sqa  f32[32768]     @ 8 KB
    //   bh   f16[131072]    @ 136 KB
    //   bl   f16[131072]
    //   ah   f16[2097152]
    //   al   f16[2097152]
    char* ws = (char*)d_ws;
    float*    sqb_ws = (float*)ws;
    float*    sqa_ws = (float*)(ws + 8192);
    _Float16* bh_ws  = (_Float16*)(ws + 8192 + 131072);
    _Float16* bl_ws  = bh_ws + BLK_ELEMS;
    _Float16* ah_ws  = bl_ws + BLK_ELEMS;
    _Float16* al_ws  = ah_ws + IMG_ELEMS;

    sqb_kernel<<<dim3(NB / 256), dim3(256), 0, stream>>>(blocks, sqb_ws);
    sqa_kernel<<<dim3((2 * LDIM) / 256), dim3(256), 0, stream>>>(img, sqa_ws);
    split_blocks_kernel<<<dim3(BLK_ELEMS / 256), dim3(256), 0, stream>>>(blocks, bh_ws, bl_ws);
    split_img_kernel<<<dim3(IMG_ELEMS / 256), dim3(256), 0, stream>>>(img, ah_ws, al_ws);

    const int total_waves = 2 * LTILES * NGROUPS;              // 32768
    const int blocks_n    = total_waves / WAVES_PER_BLOCK;     // 4096
    tile_mse_wmma_kernel<<<dim3(blocks_n), dim3(256), 0, stream>>>(
        ah_ws, al_ws, bh_ws, bl_ws, sqb_ws, sqa_ws, out);
}
